// MoELayerInt4_20633022890835
// MI455X (gfx1250) — compile-verified, matrix operand order checked by reading, not run on hardware
//
#include <hip/hip_runtime.h>

#define E_   8
#define KTOP 2
#define H_   2048
#define I_   4096
#define GS_  128
#define T_   2048

typedef _Float16 h16;
typedef __attribute__((ext_vector_type(16))) _Float16 v16h;
typedef __attribute__((ext_vector_type(8)))  _Float16 v8h;
typedef __attribute__((ext_vector_type(2)))  _Float16 v2h;
typedef __attribute__((ext_vector_type(8)))  float    v8f;

#define TM  64   // token rows per block tile
#define TN  64   // output cols per block tile
#define KC  64   // K chunk per LDS stage (2 WMMA k-steps)
#define LDT 72   // LDS row stride in halfs (144B: 16B aligned, conflict-breaking)

// K-permutation note: within every 8-column group, operand buffers hold logical
// K offsets in order {0,4,1,5,2,6,3,7} (storage pos of logical o = (o&3)*2+(o>>2)).
// Applied identically to A and B operands, so WMMA's K-reduction is unchanged,
// while int4 nibble-pair dequant results land contiguously (no v_mov_b16).
// hs is pre-permuted once into hsp; act is written permuted by ffn1's epilogue.

// Load a 16x32 f16 WMMA A/B fragment from an LDS tile stored row-major
// (row = M or N, col = K), matching the CDNA5 VGPR layout:
//   lanes 0-15 : row = lane,    elems 0-7 = K+0..7,  elems 8-15 = K+16..23
//   lanes 16-31: row = lane-16, elems 0-7 = K+8..15, elems 8-15 = K+24..31
__device__ __forceinline__ v16h frag_ld(const h16* base, int lane) {
  const int r  = lane & 15;
  const int hb = (lane >> 4) << 3;           // 0 or 8 halfs
  const h16* p = base + r * LDT + hb;
  v8h lo = *(const v8h*)(p);
  v8h hi = *(const v8h*)(p + 16);
  return __builtin_shufflevector(lo, hi, 0,1,2,3,4,5,6,7,8,9,10,11,12,13,14,15);
}

__device__ __forceinline__ v8f wmma_f16(v16h a, v16h b, v8f c) {
  return __builtin_amdgcn_wmma_f32_16x16x32_f16(false, a, false, b, (short)0, c,
                                                false, false);
}

// int4 x8 (nibble j at bits 4j) -> 8 f16 values (v-8)*s in PERMUTED order
// {v0,v4,v1,v5,v2,v6,v3,v7}: each v2h result is stored as-is (pure b32 copies).
// Bit trick: (nib | 0x6400) as f16 == 1024+v; then fma with s and -1032*s.
__device__ __forceinline__ v8h dq8p(int q, v2h sc, v2h off) {
  union U { int i; v2h h; };
  U t0, t1, t2, t3;
  t0.i = ( q        & 0x000F000F) | 0x64006400;  // {v0,v4}+1024
  t1.i = ((q >> 4)  & 0x000F000F) | 0x64006400;  // {v1,v5}+1024
  t2.i = ((q >> 8)  & 0x000F000F) | 0x64006400;  // {v2,v6}+1024
  t3.i = ((q >> 12) & 0x000F000F) | 0x64006400;  // {v3,v7}+1024
  const v2h r0 = t0.h * sc + off;                // v_pk_fma_f16
  const v2h r1 = t1.h * sc + off;
  const v2h r2 = t2.h * sc + off;
  const v2h r3 = t3.h * sc + off;
  v8h r;
  r[0] = r0[0]; r[1] = r0[1];
  r[2] = r1[0]; r[3] = r1[1];
  r[4] = r2[0]; r[5] = r2[1];
  r[6] = r3[0]; r[7] = r3[1];
  return r;
}

// Permute 8 consecutive logical halfs into the K-storage order.
__device__ __forceinline__ v8h kperm(v8h x) {
  return __builtin_shufflevector(x, x, 0, 4, 1, 5, 2, 6, 3, 7);
}

// ---------------------- one-pass hs K-permutation -------------------------
__global__ __launch_bounds__(256) void moe_permute(const h16* __restrict__ hs,
                                                   h16* __restrict__ hsp) {
  const long long off = ((long long)blockIdx.x * blockDim.x + threadIdx.x) * 8;
  if (off >= (long long)T_ * H_) return;
  *(v8h*)(hsp + off) = kperm(*(const v8h*)(hs + off));
}

// ------------------------------ gating ------------------------------------
__global__ __launch_bounds__(256) void moe_gate(const h16* __restrict__ hs,
                                                const float* __restrict__ gw,
                                                float* __restrict__ topw,
                                                int* __restrict__ topi,
                                                int* __restrict__ counts) {
  const int t = blockIdx.x * blockDim.x + threadIdx.x;
  if (t >= T_) return;
  const h16* hrow = hs + (long long)t * H_;
  float acc[E_];
#pragma unroll
  for (int e = 0; e < E_; ++e) acc[e] = 0.f;
  for (int h = 0; h < H_; h += 8) {
    const v8h x = *(const v8h*)(hrow + h);
#pragma unroll
    for (int j = 0; j < 8; ++j) {
      const float xf = (float)x[j];
#pragma unroll
      for (int e = 0; e < E_; ++e) acc[e] += xf * gw[e * H_ + h + j];  // uniform -> scalar loads
    }
  }
  float m = acc[0];
#pragma unroll
  for (int e = 1; e < E_; ++e) m = fmaxf(m, acc[e]);
  float p[E_]; float s = 0.f;
#pragma unroll
  for (int e = 0; e < E_; ++e) { p[e] = __expf(acc[e] - m); s += p[e]; }
#pragma unroll
  for (int e = 0; e < E_; ++e) p[e] /= s;
  int b0 = 0;
#pragma unroll
  for (int e = 1; e < E_; ++e) if (p[e] > p[b0]) b0 = e;
  int b1 = (b0 == 0) ? 1 : 0;
#pragma unroll
  for (int e = 0; e < E_; ++e) if (e != b0 && p[e] > p[b1]) b1 = e;
  const float w0 = p[b0], w1 = p[b1], inv = 1.f / (w0 + w1);
  topi[t * 2 + 0] = b0;       topi[t * 2 + 1] = b1;
  topw[t * 2 + 0] = w0 * inv; topw[t * 2 + 1] = w1 * inv;
  atomicAdd(&counts[b0], 1);
  atomicAdd(&counts[b1], 1);
}

__global__ void moe_scan(const int* __restrict__ counts,
                         int* __restrict__ offsets, int* __restrict__ cursors) {
  if (threadIdx.x == 0) {
    int o = 0;
    for (int e = 0; e < E_; ++e) { offsets[e] = o; cursors[e] = o; o += counts[e]; }
  }
}

__global__ __launch_bounds__(256) void moe_assign(const int* __restrict__ topi,
                                                  int* __restrict__ cursors,
                                                  int* __restrict__ slot_of,
                                                  int* __restrict__ token_of) {
  const int t = blockIdx.x * blockDim.x + threadIdx.x;
  if (t >= T_) return;
#pragma unroll
  for (int k = 0; k < KTOP; ++k) {
    const int e = topi[t * 2 + k];
    const int s = atomicAdd(&cursors[e], 1);
    slot_of[t * 2 + k] = s;
    token_of[s] = t;
  }
}

// --------------------------- FFN stage 1 ----------------------------------
// act[slot][i] = silu(hsp@w1^T + b1) * (hsp@w3^T + b3).
// Wave mapping: 8 waves, each computes 2 M-subtiles x 1 N-subtile so every
// B fragment feeds 2 WMMAs (16 ds_load_b128 -> 8 WMMAs per chunk).
// hsp arrives pre-permuted; act is written permuted (address swizzle only).
__global__ __launch_bounds__(256) void moe_ffn1(
    const h16* __restrict__ hsp,
    const int* __restrict__ w1q, const int* __restrict__ w3q,
    const h16* __restrict__ w13s, const h16* __restrict__ w13b,
    const int* __restrict__ counts, const int* __restrict__ offsets,
    const int* __restrict__ token_of, h16* __restrict__ act) {
  const int e   = blockIdx.z;
  const int cnt = counts[e];
  const int yt  = blockIdx.y;
  if (yt * TM >= cnt) return;
  const int i0    = blockIdx.x * TN;
  const int slot0 = offsets[e] + yt * TM;

  __shared__ alignas(16) h16 As [2][TM * LDT];
  __shared__ alignas(16) h16 W1s[2][TN * LDT];
  __shared__ alignas(16) h16 W3s[2][TN * LDT];

  const int tid  = threadIdx.x;
  const int lane = tid & 31;
  const int wave = tid >> 5;
  const int wm   = wave >> 2;  // 0..1 : rows wm*32 .. +31 (2 subtiles)
  const int wn   = wave & 3;   // 0..3 : cols wn*16 .. +15 (1 subtile)

  // A-tile gather mapping: 64 rows x 4 segs of 16 halfs
  const int  arow   = tid >> 2;
  const int  aseg   = (tid & 3) * 16;
  const bool avalid = (yt * TM + arow) < cnt;
  const long long tok = avalid ? (long long)token_of[slot0 + arow] : 0;

  // dequant mapping: 64 rows x 4 pairs of int32 (16 halfs each)
  const int drow = tid >> 2;
  const int dq   = (tid & 3) * 2;
  const long long wrow  = ((long long)e * I_ + (i0 + drow)) * (H_ / 8);
  const long long srow1 = ((long long)e * 2 * I_ + (i0 + drow)) * (H_ / GS_);
  const long long srow3 = ((long long)e * 2 * I_ + I_ + (i0 + drow)) * (H_ / GS_);

  // staging registers (global -> reg, later reg -> LDS)
  v8h sa0, sa1;
  int sq1[2], sq3[2];
  v2h sc1, sc3, of1, of3;

  auto load_stage = [&](int kb) {
#pragma unroll
    for (int j = 0; j < 8; ++j) { sa0[j] = (h16)0; sa1[j] = (h16)0; }
    if (avalid) {
      const h16* src = hsp + tok * H_ + kb + aseg;
      sa0 = *(const v8h*)(src);
      sa1 = *(const v8h*)(src + 8);
    }
    const float s1 = (float)w13s[srow1 + (kb / GS_)];
    const float s3 = (float)w13s[srow3 + (kb / GS_)];
    sc1[0] = (h16)s1;                 sc1[1] = sc1[0];
    sc3[0] = (h16)s3;                 sc3[1] = sc3[0];
    of1[0] = (h16)(-1032.0f * s1);    of1[1] = of1[0];
    of3[0] = (h16)(-1032.0f * s3);    of3[1] = of3[0];
#pragma unroll
    for (int u = 0; u < 2; ++u) {
      sq1[u] = w1q[wrow + kb / 8 + dq + u];
      sq3[u] = w3q[wrow + kb / 8 + dq + u];
    }
    if (kb + KC < H_) {  // prefetch K+2 chunk -> global_prefetch_b8
      __builtin_prefetch(w1q + wrow + (kb + KC) / 8 + dq, 0, 3);
      __builtin_prefetch(w3q + wrow + (kb + KC) / 8 + dq, 0, 3);
    }
  };

  auto store_stage = [&](int buf) {
    *(v8h*)&As[buf][arow * LDT + aseg]     = sa0;   // already permuted
    *(v8h*)&As[buf][arow * LDT + aseg + 8] = sa1;
#pragma unroll
    for (int u = 0; u < 2; ++u) {
      const int c0 = (dq + u) * 8;
      *(v8h*)&W1s[buf][drow * LDT + c0] = dq8p(sq1[u], sc1, of1);
      *(v8h*)&W3s[buf][drow * LDT + c0] = dq8p(sq3[u], sc3, of3);
    }
  };

  v8f acc1[2], acc3[2];   // indexed by M-subtile
#pragma unroll
  for (int n = 0; n < 2; ++n) {
#pragma unroll
    for (int j = 0; j < 8; ++j) { acc1[n][j] = 0.f; acc3[n][j] = 0.f; }
  }

  const int NK = H_ / KC;  // 32, even
  auto step = [&](int kt, int buf) {
    const bool more = (kt + 1) < NK;
    if (more) load_stage((kt + 1) * KC);   // global loads before compute

    const h16* Ab = &As[buf][(wm * 32) * LDT];
    v16h a[2][2];                           // [msub][ks]
    a[0][0] = frag_ld(Ab + 0,             lane);
    a[0][1] = frag_ld(Ab + 32,            lane);
    a[1][0] = frag_ld(Ab + 16 * LDT,      lane);
    a[1][1] = frag_ld(Ab + 16 * LDT + 32, lane);
    const int n0 = (wn * 16) * LDT;
    v16h b1[2], b3[2];                      // [ks], each reused by 2 WMMAs
    b1[0] = frag_ld(&W1s[buf][n0 + 0],  lane);
    b1[1] = frag_ld(&W1s[buf][n0 + 32], lane);
    b3[0] = frag_ld(&W3s[buf][n0 + 0],  lane);
    b3[1] = frag_ld(&W3s[buf][n0 + 32], lane);
#pragma unroll
    for (int m = 0; m < 2; ++m) {
#pragma unroll
      for (int ks = 0; ks < 2; ++ks) {
        acc1[m] = wmma_f16(a[m][ks], b1[ks], acc1[m]);
        acc3[m] = wmma_f16(a[m][ks], b3[ks], acc3[m]);
      }
    }

    if (more) store_stage(buf ^ 1);        // dequant + LDS store after compute
    __syncthreads();
  };

  load_stage(0);
  store_stage(0);
  __syncthreads();
  for (int kt = 0; kt < NK; kt += 2) {  // buf is compile-time after inlining
    step(kt, 0);
    step(kt + 1, 1);
  }

  // epilogue: bias + silu(h1)*h3, C-layout: N=lane&15, M=(lane>>4)*8 + vgpr.
  // act columns are stored K-permuted: pos(o) = (o&3)*2 + (o>>2) within 8-groups.
  const int nl = lane & 15;
  const int mh = (lane >> 4) * 8;
  const int icol = i0 + wn * 16 + nl;
  const int pcol = (icol & ~7) | (((icol & 3) << 1) | ((icol >> 2) & 1));
  const float b1 = (float)w13b[(long long)e * 2 * I_ + icol];
  const float b3 = (float)w13b[(long long)e * 2 * I_ + I_ + icol];
#pragma unroll
  for (int m = 0; m < 2; ++m) {
#pragma unroll
    for (int j = 0; j < 8; ++j) {
      const int ml = wm * 32 + m * 16 + mh + j;
      if (yt * TM + ml < cnt) {
        const float h1 = acc1[m][j] + b1;
        const float h3 = acc3[m][j] + b3;
        const float v  = h1 * (1.f / (1.f + __expf(-h1))) * h3;
        act[(long long)(slot0 + ml) * I_ + pcol] = (h16)v;
      }
    }
  }
}

// --------------------------- FFN stage 2 ----------------------------------
// outb[slot][h] = act[slot] @ w2[e]^T + b2.  act rows arrive pre-permuted, so
// A staging is a straight copy; w2 dequant uses the same permuted order.
__global__ __launch_bounds__(256) void moe_ffn2(
    const h16* __restrict__ act,
    const int* __restrict__ w2q, const h16* __restrict__ w2s,
    const h16* __restrict__ w2b,
    const int* __restrict__ counts, const int* __restrict__ offsets,
    h16* __restrict__ outb) {
  const int e   = blockIdx.z;
  const int cnt = counts[e];
  const int yt  = blockIdx.y;
  if (yt * TM >= cnt) return;
  const int h0    = blockIdx.x * TN;
  const int slot0 = offsets[e] + yt * TM;

  __shared__ alignas(16) h16 As [2][TM * LDT];
  __shared__ alignas(16) h16 W2s[2][TN * LDT];

  const int tid  = threadIdx.x;
  const int lane = tid & 31;
  const int wave = tid >> 5;
  const int wm   = wave >> 2;  // 0..1 : 2 M-subtiles
  const int wn   = wave & 3;   // 0..3 : 1 N-subtile

  const int  arow   = tid >> 2;
  const int  aseg   = (tid & 3) * 16;
  const bool avalid = (yt * TM + arow) < cnt;

  const int drow = tid >> 2;
  const int dq   = (tid & 3) * 2;
  const long long wrow = ((long long)e * H_ + (h0 + drow)) * (I_ / 8);
  const long long srow = ((long long)e * H_ + (h0 + drow)) * (I_ / GS_);

  v8h sa0, sa1;
  int sq[2];
  v2h scv, ofv;

  auto load_stage = [&](int kb) {
#pragma unroll
    for (int j = 0; j < 8; ++j) { sa0[j] = (h16)0; sa1[j] = (h16)0; }
    if (avalid) {
      const h16* src = act + (long long)(slot0 + arow) * I_ + kb + aseg;
      sa0 = *(const v8h*)(src);
      sa1 = *(const v8h*)(src + 8);
    }
    const float s = (float)w2s[srow + (kb / GS_)];
    scv[0] = (h16)s;               scv[1] = scv[0];
    ofv[0] = (h16)(-1032.0f * s);  ofv[1] = ofv[0];
#pragma unroll
    for (int u = 0; u < 2; ++u) sq[u] = w2q[wrow + kb / 8 + dq + u];
    if (kb + KC < I_)
      __builtin_prefetch(w2q + wrow + (kb + KC) / 8 + dq, 0, 3);
  };

  auto store_stage = [&](int buf) {
    *(v8h*)&As[buf][arow * LDT + aseg]     = sa0;   // already permuted
    *(v8h*)&As[buf][arow * LDT + aseg + 8] = sa1;
#pragma unroll
    for (int u = 0; u < 2; ++u)
      *(v8h*)&W2s[buf][drow * LDT + (dq + u) * 8] = dq8p(sq[u], scv, ofv);
  };

  v8f acc[2];   // indexed by M-subtile
#pragma unroll
  for (int n = 0; n < 2; ++n)
#pragma unroll
    for (int j = 0; j < 8; ++j) acc[n][j] = 0.f;

  const int NK = I_ / KC;  // 64, even
  auto step = [&](int kt, int buf) {
    const bool more = (kt + 1) < NK;
    if (more) load_stage((kt + 1) * KC);

    const h16* Ab = &As[buf][(wm * 32) * LDT];
    v16h a[2][2];
    a[0][0] = frag_ld(Ab + 0,             lane);
    a[0][1] = frag_ld(Ab + 32,            lane);
    a[1][0] = frag_ld(Ab + 16 * LDT,      lane);
    a[1][1] = frag_ld(Ab + 16 * LDT + 32, lane);
    const int n0 = (wn * 16) * LDT;
    v16h b[2];
    b[0] = frag_ld(&W2s[buf][n0 + 0],  lane);
    b[1] = frag_ld(&W2s[buf][n0 + 32], lane);
#pragma unroll
    for (int m = 0; m < 2; ++m) {
#pragma unroll
      for (int ks = 0; ks < 2; ++ks)
        acc[m] = wmma_f16(a[m][ks], b[ks], acc[m]);
    }

    if (more) store_stage(buf ^ 1);
    __syncthreads();
  };

  load_stage(0);
  store_stage(0);
  __syncthreads();
  for (int kt = 0; kt < NK; kt += 2) {
    step(kt, 0);
    step(kt + 1, 1);
  }

  const int nl = lane & 15;
  const int mh = (lane >> 4) * 8;
  const int hcol = h0 + wn * 16 + nl;
  const float bb = (float)w2b[(long long)e * H_ + hcol];
#pragma unroll
  for (int m = 0; m < 2; ++m) {
#pragma unroll
    for (int j = 0; j < 8; ++j) {
      const int ml = wm * 32 + m * 16 + mh + j;
      if (yt * TM + ml < cnt)
        outb[(long long)(slot0 + ml) * H_ + hcol] = (h16)(acc[m][j] + bb);
    }
  }
}

// ----------------------------- combine ------------------------------------
__global__ __launch_bounds__(256) void moe_final(const h16* __restrict__ outb,
                                                 const int* __restrict__ slot_of,
                                                 const float* __restrict__ topw,
                                                 h16* __restrict__ out) {
  const long long idx = (long long)blockIdx.x * blockDim.x + threadIdx.x;
  const int cpr = H_ / 8;
  const int t = (int)(idx / cpr);
  const int c = (int)(idx % cpr) * 8;
  if (t >= T_) return;
  const int   s0 = slot_of[t * 2 + 0], s1 = slot_of[t * 2 + 1];
  const float w0 = topw[t * 2 + 0],   w1 = topw[t * 2 + 1];
  const v8h a = *(const v8h*)&outb[(long long)s0 * H_ + c];
  const v8h b = *(const v8h*)&outb[(long long)s1 * H_ + c];
  v8h r;
#pragma unroll
  for (int j = 0; j < 8; ++j)
    r[j] = (h16)(w0 * (float)a[j] + w1 * (float)b[j]);
  *(v8h*)&out[(long long)t * H_ + c] = r;
}

// ------------------------------ launcher ----------------------------------
extern "C" void kernel_launch(void* const* d_in, const int* in_sizes, int n_in,
                              void* d_out, int out_size, void* d_ws, size_t ws_size,
                              hipStream_t stream) {
  (void)in_sizes; (void)n_in; (void)out_size; (void)ws_size;
  const h16*   hs   = (const h16*)  d_in[0];
  const float* gw   = (const float*)d_in[1];
  const int*   w1q  = (const int*)  d_in[2];
  const int*   w3q  = (const int*)  d_in[3];
  const int*   w2q  = (const int*)  d_in[4];
  const h16*   w13s = (const h16*)  d_in[5];
  const h16*   w2s  = (const h16*)  d_in[6];
  const h16*   w13b = (const h16*)  d_in[7];
  const h16*   w2b  = (const h16*)  d_in[8];
  h16*         out  = (h16*)d_out;

  char* p = (char*)d_ws;
  float* topw     = (float*)p; p += (size_t)T_ * 2 * 4;   // 16 KB
  int*   topi     = (int*)p;   p += (size_t)T_ * 2 * 4;   // 16 KB
  int*   slot_of  = (int*)p;   p += (size_t)T_ * 2 * 4;   // 16 KB
  int*   token_of = (int*)p;   p += (size_t)T_ * 2 * 4;   // 16 KB
  int*   counts   = (int*)p;   p += 256;
  int*   offsets  = (int*)p;   p += 256;
  int*   cursors  = (int*)p;   p += 256;
  h16*   hsp      = (h16*)p;   p += (size_t)T_ * H_ * 2;         //  8 MB
  h16*   act      = (h16*)p;   p += (size_t)T_ * KTOP * I_ * 2;  // 32 MB
  h16*   outb     = (h16*)p;                                      // 16 MB

  hipMemsetAsync(counts, 0, E_ * sizeof(int), stream);

  moe_permute<<<(T_ * (H_ / 8)) / 256, 256, 0, stream>>>(hs, hsp);
  moe_gate  <<<T_ / 256, 256, 0, stream>>>(hs, gw, topw, topi, counts);
  moe_scan  <<<1, 32, 0, stream>>>(counts, offsets, cursors);
  moe_assign<<<T_ / 256, 256, 0, stream>>>(topi, cursors, slot_of, token_of);

  dim3 g1(I_ / TN, T_ / TM, E_);   // early-exit on count[e]
  moe_ffn1<<<g1, 256, 0, stream>>>(hsp, w1q, w3q, w13s, w13b,
                                   counts, offsets, token_of, act);

  dim3 g2(H_ / TN, T_ / TM, E_);
  moe_ffn2<<<g2, 256, 0, stream>>>(act, w2q, w2s, w2b, counts, offsets, outb);

  moe_final<<<(T_ * (H_ / 8)) / 256, 256, 0, stream>>>(outb, slot_of, topw, out);
}
